// DeformableConv2d_18124761989871
// MI455X (gfx1250) — compile-verified, matrix operand order checked by reading, not run on hardware
//
#include <hip/hip_runtime.h>

// ---------------------------------------------------------------------------
// Deformable Conv2d for gfx1250 (MI455X, wave32, WMMA).
//   1) prep_main_w : conv_w  fp32 -> bf16 B-fragments (32x16 KxN tiles)
//   2) prep_off_w  : offset_w fp32 -> fp32 4x16 B-fragments, K-order t*128+c
//   3) offconv     : offset conv via v_wmma_f32_16x16x4_f32 (exact fp32)
//   4) dconv       : 64-pixel M tile: bilinear-sample -> LDS bf16 A-frags ->
//                    v_wmma_f32_16x16x32_bf16 with 4x B-fragment reuse;
//                    LDS-transposed global_store_b128 epilogue
// ---------------------------------------------------------------------------

typedef __attribute__((ext_vector_type(16))) __bf16 v16bf;
typedef __attribute__((ext_vector_type(8)))  float  v8f;
typedef __attribute__((ext_vector_type(2)))  float  v2f;

#define Bn     8
#define Cn     128
#define Hn     96
#define Wn     96
#define On     128
#define HW     (Hn * Wn)        // 9216
#define K2n    9
#define OFFC   18               // 2*K*K offset channels
#define NTILES (HW / 16)        // 576 16-pixel tiles (offconv)
#define MT     4                // M-subtiles per dconv block
#define MPIX   (MT * 16)        // 64 pixels per dconv block
#define MTILES (HW / MPIX)      // 144 dconv tiles per image

// d_ws layout (bytes):
//   [0)                        offsets  float[B*18*HW]      = 5,308,416 B
//   [OFFS_BYTES)               main-W   bf16 [9*4*8*32*16]  =   294,912 B
//   [OFFS_BYTES+WMAIN_BYTES)   off-W    fp32 [288*2*32*2]   =   147,456 B
#define OFFS_ELEMS  (Bn * OFFC * HW)
#define OFFS_BYTES  ((size_t)OFFS_ELEMS * 4)
#define WMAIN_ELEMS (K2n * 4 * 8 * 32 * 16)   // 147456 bf16
#define WMAIN_BYTES ((size_t)WMAIN_ELEMS * 2)
#define WOFF_ELEMS  (288 * 2 * 32 * 2)        // 36864 floats

__device__ __forceinline__ unsigned short f32_to_bf16_rne(float f) {
  unsigned u = __float_as_uint(f);
  unsigned r = u + 0x7FFFu + ((u >> 16) & 1u);   // round-to-nearest-even
  return (unsigned short)(r >> 16);
}

// ---------------------------------------------------------------------------
// bf16 B-fragment layout (32x16 KxN, mirrors documented 16-bit A 16x32):
//   n = lane&15, half = lane>>4, element e: k = (e>=8?16:0) + (e&7) + 8*half
// Index: ((((t*4+cb)*8+nb)*32 + lane)*16 + e)   (lane block contiguous, 32 B)
// ---------------------------------------------------------------------------
__global__ __launch_bounds__(256) void prep_main_w(const float* __restrict__ w,
                                                   unsigned short* __restrict__ wf) {
  int idx  = blockIdx.x * 256 + threadIdx.x;     // [0, 147456)
  int e    = idx & 15;
  int lane = (idx >> 4) & 31;
  int nb   = (idx >> 9) & 7;
  int cb   = (idx >> 12) & 3;
  int t    = idx >> 14;                          // [0,9)
  int half = lane >> 4;
  int n    = lane & 15;
  int k    = ((e >= 8) ? 16 : 0) + (e & 7) + 8 * half;
  int oc   = nb * 16 + n;
  int c    = cb * 32 + k;
  wf[idx]  = f32_to_bf16_rne(w[((size_t)oc * Cn + c) * K2n + t]);
}

// fp32 4x16 B-fragment (v_wmma_f32_16x16x4_f32): 2 floats/lane,
//   n = lane&15, half = lane>>4, k = v + 2*half  (v in {0,1})
// K-order: k_global = t*128 + c  (tap-major => division-free offconv loop)
__global__ __launch_bounds__(256) void prep_off_w(const float* __restrict__ w,
                                                  float* __restrict__ wf) {
  int idx  = blockIdx.x * 256 + threadIdx.x;     // [0, 36864)
  int v    = idx & 1;
  int lane = (idx >> 1) & 31;
  int nb   = (idx >> 6) & 1;
  int s    = idx >> 7;                           // [0, 288)
  int half = lane >> 4;
  int n    = lane & 15;
  int k    = s * 4 + v + 2 * half;               // [0, 1152)
  int t    = k >> 7;                             // tap
  int c    = k & 127;                            // channel
  int oc   = nb * 16 + n;
  float val = 0.f;
  if (oc < OFFC) val = w[((size_t)oc * Cn + c) * K2n + t];
  wf[idx] = val;
}

// ---------------------------------------------------------------------------
// Offset conv: one wave per 16-pixel tile, exact fp32 WMMA. Tap-outer loop,
// branch-free inner 32 K-steps with immediate-offset channel loads.
// ---------------------------------------------------------------------------
__global__ __launch_bounds__(256) void offconv(const float* __restrict__ x,
                                               const float* __restrict__ wf,
                                               const float* __restrict__ bias,
                                               float* __restrict__ offs) {
  int wid  = (blockIdx.x * 256 + threadIdx.x) >> 5;  // global wave id [0,4608)
  int lane = threadIdx.x & 31;
  int b    = wid / NTILES;
  int tile = wid - b * NTILES;
  int half = lane >> 4;
  int nlo  = lane & 15;
  int pix  = tile * 16 + nlo;           // A row m = lane&15
  int py   = pix / Wn, px = pix - (pix / Wn) * Wn;
  const float* xb = x + (size_t)b * Cn * HW;

  v8f acc0, acc1;
  float b0 = bias[nlo];
  float b1 = (nlo < 2) ? bias[16 + nlo] : 0.f;
#pragma unroll
  for (int r = 0; r < 8; ++r) { acc0[r] = b0; acc1[r] = b1; }

  for (int t = 0; t < K2n; ++t) {
    int ky = t / 3, kx = t - ky * 3;
    int yy = py + ky - 1, xx = px + kx - 1;
    float fv = (yy >= 0 && yy < Hn && xx >= 0 && xx < Wn) ? 1.f : 0.f;
    int cy = min(max(yy, 0), Hn - 1), cx = min(max(xx, 0), Wn - 1);
    const float* base = xb + cy * Wn + cx;       // + c*HW per channel
    const float* wb   = wf + (size_t)t * 32 * 2 * 32 * 2;
#pragma unroll 8
    for (int cs = 0; cs < 32; ++cs) {
      int c0 = cs * 4 + 2 * half;                // lane's K pair
      v2f a;
      a[0] = fv * base[(size_t)c0 * HW];
      a[1] = fv * base[(size_t)(c0 + 1) * HW];
      v2f w0 = *(const v2f*)&wb[((size_t)(cs * 2 + 0) * 32 + lane) * 2];
      v2f w1 = *(const v2f*)&wb[((size_t)(cs * 2 + 1) * 32 + lane) * 2];
      acc0 = __builtin_amdgcn_wmma_f32_16x16x4_f32(false, a, false, w0, (short)0,
                                                   acc0, false, false);
      acc1 = __builtin_amdgcn_wmma_f32_16x16x4_f32(false, a, false, w1, (short)0,
                                                   acc1, false, false);
    }
  }
#pragma unroll
  for (int r = 0; r < 8; ++r) {
    int mm   = r + 8 * half;            // D: VGPR r -> M = r + 8*half
    int opix = tile * 16 + mm;
    offs[((size_t)b * OFFC + nlo) * HW + opix] = acc0[r];
    if (nlo < 2)
      offs[((size_t)b * OFFC + 16 + nlo) * HW + opix] = acc1[r];
  }
}

// ---------------------------------------------------------------------------
// Main deformable conv. Block = (batch, 64-pixel tile), 256 threads / 8 waves.
// Phase 1: 1152 sampler units (64 pix x 9 taps x 2 channel-halves) strided
//          over all threads; bf16 results pre-swizzled into LDS A-fragments.
// Phase 2: wave w owns oc block w, 4 accumulators; each B fragment loaded
//          once per (tap, cblock) and reused across 4 M-subtiles (4x less
//          L2 weight traffic). 144 x v_wmma_f32_16x16x32_bf16 per wave.
// Phase 3: D tiles transposed through (reused) LDS -> global_store_b128.
// ---------------------------------------------------------------------------
#define STP 20                           // staging row stride: 16B-aligned rows
__global__ __launch_bounds__(256) void dconv(const float* __restrict__ x,
                                             const float* __restrict__ offs,
                                             const unsigned short* __restrict__ wfrag,
                                             const float* __restrict__ bias,
                                             float* __restrict__ out) {
  __shared__ unsigned short AF[MT * K2n * 4 * 32 * 16];   // 147,456 B
  int tile = blockIdx.x;
  int b    = blockIdx.y;
  int tid  = threadIdx.x;
  const float* xb = x + (size_t)b * Cn * HW;

  for (int u = tid; u < MPIX * K2n * 2; u += 256) {  // 1152 sampler units
    int chalf = u & 1;
    int v64   = u >> 1;                 // [0, 576)
    int p64   = v64 & 63;               // pixel in 64-tile
    int t     = v64 >> 6;               // tap
    int mt    = p64 >> 4;               // M-subtile
    int p     = p64 & 15;               // pixel in subtile
    int pix   = tile * MPIX + p64;
    int oy    = pix / Wn, ox = pix - (pix / Wn) * Wn;
    float dy  = offs[((size_t)b * OFFC + 2 * t + 0) * HW + pix];
    float dx  = offs[((size_t)b * OFFC + 2 * t + 1) * HW + pix];
    float fy  = (float)(oy - 1 + t / 3) + dy;
    float fx  = (float)(ox - 1 + (t - (t / 3) * 3)) + dx;
    float y0f = floorf(fy), x0f = floorf(fx);
    float wy = fy - y0f,    wx = fx - x0f;
    int y0 = (int)y0f, x0 = (int)x0f, y1 = y0 + 1, x1 = x0 + 1;
    bool vy0 = (y0 >= 0) & (y0 < Hn), vy1 = (y1 >= 0) & (y1 < Hn);
    bool vx0 = (x0 >= 0) & (x0 < Wn), vx1 = (x1 >= 0) & (x1 < Wn);
    int cy0 = min(max(y0, 0), Hn - 1), cy1 = min(max(y1, 0), Hn - 1);
    int cx0 = min(max(x0, 0), Wn - 1), cx1 = min(max(x1, 0), Wn - 1);
    float w00 = (1.f - wy) * (1.f - wx) * (float)(vy0 && vx0);
    float w01 = (1.f - wy) * wx         * (float)(vy0 && vx1);
    float w10 = wy * (1.f - wx)         * (float)(vy1 && vx0);
    float w11 = wy * wx                 * (float)(vy1 && vx1);
    int i00 = cy0 * Wn + cx0, i01 = cy0 * Wn + cx1;
    int i10 = cy1 * Wn + cx0, i11 = cy1 * Wn + cx1;
    const float* xc = xb + (size_t)(chalf * 64) * HW;
#pragma unroll 4
    for (int cc = 0; cc < 64; ++cc) {
      int c = chalf * 64 + cc;
      float v = w00 * xc[(size_t)cc * HW + i00] + w01 * xc[(size_t)cc * HW + i01]
              + w10 * xc[(size_t)cc * HW + i10] + w11 * xc[(size_t)cc * HW + i11];
      // swizzle (mt, t, c, p) into bf16 A-fragment order (A is 16x32 MxK):
      int cb  = c >> 5;
      int k   = c & 31;
      int eb  = (k >= 16) ? 8 : 0;
      int rem = k & 15;
      int hlf = rem >> 3;
      int e   = eb + (rem & 7);
      AF[((mt * (K2n * 4) + t * 4 + cb) * 32 + (p + 16 * hlf)) * 16 + e] =
          f32_to_bf16_rne(v);
    }
  }
  __syncthreads();

  int wave = tid >> 5;                  // oc block
  int lane = tid & 31;
  int nlo  = lane & 15;
  int half = lane >> 4;
  v8f acc[MT];
  float bs = bias[wave * 16 + nlo];
#pragma unroll
  for (int mt = 0; mt < MT; ++mt)
#pragma unroll
    for (int r = 0; r < 8; ++r) acc[mt][r] = bs;

  for (int t = 0; t < K2n; ++t) {
#pragma unroll
    for (int cb = 0; cb < 4; ++cb) {
      // one B-fragment load, reused for 4 M-subtiles
      v16bf w = *(const v16bf*)&wfrag[(((size_t)(t * 4 + cb) * 8 + wave) * 32 + lane) * 16];
#pragma unroll
      for (int mt = 0; mt < MT; ++mt) {
        v16bf a = *(const v16bf*)&AF[((mt * (K2n * 4) + t * 4 + cb) * 32 + lane) * 16];
        acc[mt] = __builtin_amdgcn_wmma_f32_16x16x32_bf16(false, a, false, w, (short)0,
                                                          acc[mt], false, false);
      }
    }
  }

  // Epilogue: reuse AF space as fp32 staging (all A reads are done).
  __syncthreads();
  float* stg = (float*)AF;              // 32 tiles x 16 rows x STP floats = 40,960 B
#pragma unroll
  for (int mt = 0; mt < MT; ++mt) {
    float* st = &stg[(wave * MT + mt) * 16 * STP];
    *(float4*)&st[nlo * STP + 8 * half]     = make_float4(acc[mt][0], acc[mt][1], acc[mt][2], acc[mt][3]);
    *(float4*)&st[nlo * STP + 8 * half + 4] = make_float4(acc[mt][4], acc[mt][5], acc[mt][6], acc[mt][7]);
  }
  __syncthreads();
  int ocl = lane >> 1, pof = (lane & 1) * 8;       // lane -> (oc row, pixel 8-group)
#pragma unroll
  for (int mt = 0; mt < MT; ++mt) {
    const float* st = &stg[(wave * MT + mt) * 16 * STP];
    float4 o0 = *(const float4*)&st[ocl * STP + pof];
    float4 o1 = *(const float4*)&st[ocl * STP + pof + 4];
    float* op = &out[((size_t)b * On + wave * 16 + ocl) * HW + tile * MPIX + mt * 16 + pof];
    *(float4*)&op[0] = o0;
    *(float4*)&op[4] = o1;
  }
}

extern "C" void kernel_launch(void* const* d_in, const int* in_sizes, int n_in,
                              void* d_out, int out_size, void* d_ws, size_t ws_size,
                              hipStream_t stream) {
  const float* x        = (const float*)d_in[0];
  const float* offset_w = (const float*)d_in[1];
  const float* offset_b = (const float*)d_in[2];
  const float* conv_w   = (const float*)d_in[3];
  const float* conv_b   = (const float*)d_in[4];
  float* out = (float*)d_out;

  float*          offs  = (float*)d_ws;
  unsigned short* wmain = (unsigned short*)((char*)d_ws + OFFS_BYTES);
  float*          woff  = (float*)((char*)d_ws + OFFS_BYTES + WMAIN_BYTES);

  prep_main_w<<<WMAIN_ELEMS / 256, 256, 0, stream>>>(conv_w, wmain);
  prep_off_w <<<WOFF_ELEMS  / 256, 256, 0, stream>>>(offset_w, woff);
  offconv    <<<(Bn * NTILES) / 8, 256, 0, stream>>>(x, woff, offset_b, offs);
  dconv      <<<dim3(MTILES, Bn),  256, 0, stream>>>(x, offs, wmain, conv_b, out);
}
